// SelfAttention_29729763623614
// MI455X (gfx1250) — matrix operand order repaired and checked by reading, not compiled, not run on hardware
//
#include <hip/hip_runtime.h>

#define D_MODEL 1024
#define SEQ     2048
#define NBATCH  2
#define NHEADS  16
#define HD      64
#define NROWS   (NBATCH * SEQ)        // 4096 token rows

typedef __attribute__((ext_vector_type(16))) __bf16 v16bf;
typedef __attribute__((ext_vector_type(8)))  __bf16 v8bf;
typedef __attribute__((ext_vector_type(8)))  float  v8f;

// ---------- helpers ----------
static __device__ inline __bf16 f2bf(float f) {
  union { float f; unsigned u; } v; v.f = f;
  unsigned r = v.u + 0x7FFFu + ((v.u >> 16) & 1u);   // round-to-nearest-even
  unsigned short h = (unsigned short)(r >> 16);
  return __builtin_bit_cast(__bf16, h);
}

static __device__ inline v16bf cat8(v8bf lo, v8bf hi) {
  return __builtin_shufflevector(lo, hi, 0,1,2,3,4,5,6,7,8,9,10,11,12,13,14,15);
}

static __device__ inline v8f wmma_bf16(v16bf a, v16bf b, v8f c) {
  // D = A(16x32 bf16) x B(32x16 bf16) + C(16x16 f32); NEG must be 0 for bf16 A/B.
  return __builtin_amdgcn_wmma_f32_16x16x32_bf16(false, a, false, b, (short)0, c,
                                                 false, false);
}

// ---------- weight convert + transpose: wt[n][k] = (bf16) w[k][n] ----------
__global__ __launch_bounds__(256) void transpose_bf16_kernel(
    const float* __restrict__ w, __bf16* __restrict__ wt, int K, int N) {
  size_t idx = (size_t)blockIdx.x * 256 + threadIdx.x;
  if (idx < (size_t)K * N) {
    int kk = (int)(idx / N);
    int j  = (int)(idx % N);
    wt[(size_t)j * K + kk] = f2bf(w[idx]);
  }
}

// ---------- layernorm over rows of D_MODEL ----------
static __device__ inline void st_val(float* p, float v)  { *p = v; }
static __device__ inline void st_val(__bf16* p, float v) { *p = f2bf(v); }

template <typename OUT>
__global__ __launch_bounds__(256) void ln_kernel(const float* __restrict__ x,
                                                 const float* __restrict__ w,
                                                 const float* __restrict__ b,
                                                 OUT* __restrict__ out) {
  const int row = blockIdx.x;
  const int tid = threadIdx.x;
  const int wave = tid >> 5, lane = tid & 31;
  const float* xr = x + (size_t)row * D_MODEL;
  float v[4];
  #pragma unroll
  for (int i = 0; i < 4; ++i) v[i] = xr[i * 256 + tid];

  float s = v[0] + v[1] + v[2] + v[3];
  #pragma unroll
  for (int off = 16; off; off >>= 1) s += __shfl_xor(s, off, 32);
  __shared__ float red[8];
  if (lane == 0) red[wave] = s;
  __syncthreads();
  float tot = 0.0f;
  #pragma unroll
  for (int i = 0; i < 8; ++i) tot += red[i];
  const float mu = tot * (1.0f / (float)D_MODEL);

  float q = 0.0f;
  #pragma unroll
  for (int i = 0; i < 4; ++i) { float d = v[i] - mu; q += d * d; }
  #pragma unroll
  for (int off = 16; off; off >>= 1) q += __shfl_xor(q, off, 32);
  __syncthreads();
  if (lane == 0) red[wave] = q;
  __syncthreads();
  float qt = 0.0f;
  #pragma unroll
  for (int i = 0; i < 8; ++i) qt += red[i];
  const float rstd = rsqrtf(qt * (1.0f / (float)D_MODEL) + 1e-5f);

  #pragma unroll
  for (int i = 0; i < 4; ++i) {
    int idx = i * 256 + tid;
    st_val(out + (size_t)row * D_MODEL + idx, (v[i] - mu) * rstd * w[idx] + b[idx]);
  }
}

// ---------- 16x64 WMMA tile core: C = A[M,K] @ Bt[N,K]^T ----------
static __device__ inline void mma_16x64(const __bf16* __restrict__ A, int lda,
                                        const __bf16* __restrict__ Bt, int ldb,
                                        int row0, int col0, int kdim,
                                        v8f (&acc)[4]) {
  const int lane = (int)(threadIdx.x & 31u);
  const int m = lane & 15, half = lane >> 4;
  const __bf16* arow = A + (size_t)(row0 + m) * lda;
  const __bf16* b0   = Bt + (size_t)(col0 + m) * ldb + half * 16;
  #pragma unroll 2
  for (int kb = 0; kb < kdim; kb += 32) {
    v8bf lo = *(const v8bf*)(arow + kb + half * 16);
    v8bf hi = *(const v8bf*)(arow + kb + half * 16 + 8);
    v16bf a = cat8(lo, hi);
    #pragma unroll
    for (int j = 0; j < 4; ++j) {
      v16bf b = *(const v16bf*)(b0 + (size_t)j * 16 * ldb + kb);
      acc[j] = wmma_bf16(a, b, acc[j]);
    }
  }
}

// ---------- QKV projection: xn[4096,1024] @ W[1024,3072] -> Q,K,Vt ----------
__global__ __launch_bounds__(256) void qkv_gemm_kernel(
    const __bf16* __restrict__ xn, const __bf16* __restrict__ wt,
    const float* __restrict__ bias,
    __bf16* __restrict__ Q, __bf16* __restrict__ K, __bf16* __restrict__ Vt) {
  const int NT = (3 * D_MODEL) / 64;                 // 48 column tiles
  const int tile = blockIdx.x * 8 + (threadIdx.x >> 5);
  const int mt = tile / NT, nt = tile % NT;
  const int row0 = mt * 16, col0 = nt * 64;
  v8f acc[4] = {};
  mma_16x64(xn, D_MODEL, wt, D_MODEL, row0, col0, D_MODEL, acc);

  const int lane = (int)(threadIdx.x & 31u);
  const int m = lane & 15, half = lane >> 4;
  #pragma unroll
  for (int j = 0; j < 4; ++j) {
    const int col = col0 + j * 16 + m;
    const int which = col >> 10;                     // 0=q 1=k 2=v
    const int within = col & (D_MODEL - 1);
    const int head = within >> 6, dim = within & 63;
    const float bv = bias[col];
    #pragma unroll
    for (int r = 0; r < 8; ++r) {
      const int row = row0 + r + 8 * half;
      const int nb = row >> 11, t = row & (SEQ - 1);
      const int nh = nb * NHEADS + head;
      const float val = acc[j][r] + bv;
      if (which == 0)
        Q[((size_t)nh * SEQ + t) * HD + dim] = f2bf(val * 0.125f);  // 1/sqrt(64)
      else if (which == 1)
        K[((size_t)nh * SEQ + t) * HD + dim] = f2bf(val);
      else
        Vt[((size_t)nh * HD + dim) * SEQ + t] = f2bf(val);
    }
  }
}

// ---------- flash attention: 1 wave = one 16-row t-tile of one (n,h) ----------
__global__ __launch_bounds__(32) void attn_kernel(
    const __bf16* __restrict__ Q, const __bf16* __restrict__ K,
    const __bf16* __restrict__ Vt, const int* __restrict__ mask,
    __bf16* __restrict__ out) {
  __shared__ __attribute__((aligned(32))) __bf16 plds[16][32];
  const int lane = (int)threadIdx.x;
  const int m = lane & 15, half = lane >> 4;
  const int ttile = blockIdx.x;                      // 0..127
  const int nh = blockIdx.y;                         // 0..31
  const int nb = nh >> 4, head = nh & 15;

  const __bf16* Qb = Q  + (size_t)nh * SEQ * HD;
  const __bf16* Kb = K  + (size_t)nh * SEQ * HD;
  const __bf16* Vb = Vt + (size_t)nh * HD * SEQ;
  const int* mb = mask + nb * SEQ;

  // Q A-fragments for the two K-steps of hd=64 (resident for whole loop)
  const __bf16* qrow = Qb + (size_t)(ttile * 16 + m) * HD;
  v16bf aq[2];
  #pragma unroll
  for (int ks = 0; ks < 2; ++ks) {
    v8bf lo = *(const v8bf*)(qrow + ks * 32 + half * 16);
    v8bf hi = *(const v8bf*)(qrow + ks * 32 + half * 16 + 8);
    aq[ks] = cat8(lo, hi);
  }

  v8f o0 = {}, o1 = {}, o2 = {}, o3 = {};
  float mrow[8], lrow[8];
  #pragma unroll
  for (int r = 0; r < 8; ++r) { mrow[r] = -3.0e38f; lrow[r] = 0.0f; }

  for (int tc = 0; tc < SEQ; tc += 32) {
    const __bf16* k0 = Kb + (size_t)(tc + m) * HD + half * 16;
    const __bf16* k1 = k0 + 16 * HD;
    __builtin_prefetch(k0 + 32 * HD, 0, 0);          // next K chunk -> global_prefetch
    v8f s0 = {}, s1 = {};
    #pragma unroll
    for (int ks = 0; ks < 2; ++ks) {
      s0 = wmma_bf16(aq[ks], *(const v16bf*)(k0 + ks * 32), s0);
      s1 = wmma_bf16(aq[ks], *(const v16bf*)(k1 + ks * 32), s1);
    }

    // key padding mask (per column)
    const bool ok0 = mb[tc + m] != 0;
    const bool ok1 = mb[tc + 16 + m] != 0;
    #pragma unroll
    for (int r = 0; r < 8; ++r) {
      if (!ok0) s0[r] = -1.0e30f;
      if (!ok1) s1[r] = -1.0e30f;
    }

    // online softmax (rows live across the 16-lane half-groups)
    float alpha[8];
    #pragma unroll
    for (int r = 0; r < 8; ++r) {
      float v = fmaxf(s0[r], s1[r]);
      #pragma unroll
      for (int off = 8; off; off >>= 1) v = fmaxf(v, __shfl_xor(v, off, 32));
      const float mn = fmaxf(mrow[r], v);
      alpha[r] = __expf(mrow[r] - mn);
      mrow[r] = mn;
      const float p0 = __expf(s0[r] - mn);
      const float p1 = __expf(s1[r] - mn);
      s0[r] = p0; s1[r] = p1;
      float rs = p0 + p1;
      #pragma unroll
      for (int off = 8; off; off >>= 1) rs += __shfl_xor(rs, off, 32);
      lrow[r] = lrow[r] * alpha[r] + rs;
    }
    #pragma unroll
    for (int r = 0; r < 8; ++r) {
      o0[r] *= alpha[r]; o1[r] *= alpha[r];
      o2[r] *= alpha[r]; o3[r] *= alpha[r];
    }

    // repack P: C-layout -> LDS -> A-layout (16x32 bf16)
    __syncthreads();
    #pragma unroll
    for (int r = 0; r < 8; ++r) {
      plds[r + 8 * half][m]      = f2bf(s0[r]);
      plds[r + 8 * half][16 + m] = f2bf(s1[r]);
    }
    __syncthreads();
    v16bf ap = cat8(*(const v8bf*)&plds[m][half * 16],
                    *(const v8bf*)&plds[m][half * 16 + 8]);

    // O += P @ V   (Vt rows are contiguous over T)
    const __bf16* vrow = Vb + (size_t)m * SEQ + tc + half * 16;
    __builtin_prefetch(vrow + 32, 0, 0);
    o0 = wmma_bf16(ap, *(const v16bf*)(vrow),                    o0);
    o1 = wmma_bf16(ap, *(const v16bf*)(vrow + (size_t)16 * SEQ), o1);
    o2 = wmma_bf16(ap, *(const v16bf*)(vrow + (size_t)32 * SEQ), o2);
    o3 = wmma_bf16(ap, *(const v16bf*)(vrow + (size_t)48 * SEQ), o3);
  }

  #pragma unroll
  for (int r = 0; r < 8; ++r) {
    const float inv = 1.0f / lrow[r];
    const int row = ttile * 16 + r + 8 * half;
    __bf16* orow = out + (size_t)(nb * SEQ + row) * D_MODEL + head * HD + m;
    orow[0]  = f2bf(o0[r] * inv);
    orow[16] = f2bf(o1[r] * inv);
    orow[32] = f2bf(o2[r] * inv);
    orow[48] = f2bf(o3[r] * inv);
  }
}

// ---------- output projection: attn[4096,1024] @ out_w[1024,1024] -> y f32 ----
__global__ __launch_bounds__(256) void out_gemm_kernel(
    const __bf16* __restrict__ attn, const __bf16* __restrict__ wt,
    const float* __restrict__ bias, float* __restrict__ y) {
  const int NT = D_MODEL / 64;                       // 16
  const int tile = blockIdx.x * 8 + (threadIdx.x >> 5);
  const int mt = tile / NT, nt = tile % NT;
  const int row0 = mt * 16, col0 = nt * 64;
  v8f acc[4] = {};
  mma_16x64(attn, D_MODEL, wt, D_MODEL, row0, col0, D_MODEL, acc);

  const int lane = (int)(threadIdx.x & 31u);
  const int m = lane & 15, half = lane >> 4;
  #pragma unroll
  for (int j = 0; j < 4; ++j) {
    const int col = col0 + j * 16 + m;
    const float bv = bias[col];
    #pragma unroll
    for (int r = 0; r < 8; ++r) {
      const int row = row0 + r + 8 * half;
      y[(size_t)row * D_MODEL + col] = acc[j][r] + bv;
    }
  }
}

// ---------- host launch ----------
extern "C" void kernel_launch(void* const* d_in, const int* in_sizes, int n_in,
                              void* d_out, int out_size, void* d_ws, size_t ws_size,
                              hipStream_t stream) {
  const float* x       = (const float*)d_in[0];
  const int*   pmask   = (const int*)  d_in[1];
  const float* ln_in_w = (const float*)d_in[2];
  const float* ln_in_b = (const float*)d_in[3];
  const float* qkv_w   = (const float*)d_in[4];
  const float* qkv_b   = (const float*)d_in[5];
  const float* out_w   = (const float*)d_in[6];
  const float* out_b   = (const float*)d_in[7];
  const float* ln_o_w  = (const float*)d_in[8];
  const float* ln_o_b  = (const float*)d_in[9];
  float* out = (float*)d_out;

  // workspace layout (needs 64 MB)
  char* ws = (char*)d_ws;
  const size_t MB = 1024ull * 1024ull;
  __bf16* xn     = (__bf16*)(ws +  0 * MB);   // 4096x1024 bf16        (8 MB)
  __bf16* qkv_wt = (__bf16*)(ws +  8 * MB);   // 3072x1024 bf16        (6 MB)
  __bf16* out_wt = (__bf16*)(ws + 14 * MB);   // 1024x1024 bf16        (2 MB)
  __bf16* Qb     = (__bf16*)(ws + 16 * MB);   // [32][2048][64] bf16   (8 MB)
  __bf16* Kb     = (__bf16*)(ws + 24 * MB);   // [32][2048][64] bf16   (8 MB)
  __bf16* Vtb    = (__bf16*)(ws + 32 * MB);   // [32][64][2048] bf16   (8 MB)
  __bf16* attn   = (__bf16*)(ws + 40 * MB);   // 4096x1024 bf16        (8 MB)
  float*  y      = (float*) (ws + 48 * MB);   // 4096x1024 f32        (16 MB)

  // 1) weights -> bf16, N-major (transposed)
  transpose_bf16_kernel<<<(D_MODEL * 3 * D_MODEL + 255) / 256, 256, 0, stream>>>(
      qkv_w, qkv_wt, D_MODEL, 3 * D_MODEL);
  transpose_bf16_kernel<<<(D_MODEL * D_MODEL + 255) / 256, 256, 0, stream>>>(
      out_w, out_wt, D_MODEL, D_MODEL);

  // 2) input layernorm -> bf16
  ln_kernel<__bf16><<<NROWS, 256, 0, stream>>>(x, ln_in_w, ln_in_b, xn);

  // 3) QKV projection (WMMA)
  qkv_gemm_kernel<<<(NROWS / 16) * ((3 * D_MODEL) / 64) / 8, 256, 0, stream>>>(
      xn, qkv_wt, qkv_b, Qb, Kb, Vtb);

  // 4) flash attention (WMMA)
  attn_kernel<<<dim3(SEQ / 16, NBATCH * NHEADS), 32, 0, stream>>>(
      Qb, Kb, Vtb, pmask, attn);

  // 5) output projection (WMMA)
  out_gemm_kernel<<<(NROWS / 16) * (D_MODEL / 64) / 8, 256, 0, stream>>>(
      attn, out_wt, out_b, y);

  // 6) output layernorm -> f32 result
  ln_kernel<float><<<NROWS, 256, 0, stream>>>(y, ln_o_w, ln_o_b, out);

  (void)in_sizes; (void)n_in; (void)out_size; (void)ws_size;
}
